// Model_3D_77206332113175
// MI455X (gfx1250) — matrix operand order repaired and checked by reading, not compiled, not run on hardware
//
#include <hip/hip_runtime.h>
#include <hip/hip_bf16.h>

typedef _Float16 f16;
typedef _Float16 v16h __attribute__((ext_vector_type(16)));
typedef float    v8f  __attribute__((ext_vector_type(8)));
typedef float    v2f  __attribute__((ext_vector_type(2)));

// tap offset inside a padded (Dp,Hp,Wp) volume for implicit-GEMM k = ic*27 + kd*9 + kh*3 + kw
__device__ constexpr int koff_of(int k, int HpWp, int Wp, int Vol) {
    return (k / 27) * Vol + ((k % 27) / 9) * HpWp + (((k % 27) % 9) / 3) * Wp + ((k % 27) % 3);
}

// ---------------------------------------------------------------------------
// Conv3d 3x3x3 pad1 stride1 + ReLU, implicit GEMM on v_wmma_f32_16x16x32_f16.
// Input is pre-padded f16 (halo zeros): fully unrolled K loop with constant
// tap offsets; only lane-dependence is the ISA's A-layout half split (cndmask).
// ---------------------------------------------------------------------------
template <int IC, int OC, int D, int H, int W>
__global__ __launch_bounds__(256) void conv3d_k(
    const f16* __restrict__ xpad, const float* __restrict__ w,
    const float* __restrict__ bias, float* __restrict__ y, int B)
{
    constexpr int K    = IC * 27;
    constexpr int Kpad = (K + 31) / 32 * 32;
    constexpr int Dp = D + 2, Hp = H + 2, Wp = W + 2;
    constexpr int HpWp = Hp * Wp;
    constexpr int Vol  = Dp * HpWp;
    constexpr int spatial = D * H * W;
    __shared__ f16 wlds[16 * Kpad];

    const int tid   = threadIdx.x;
    const int ntile = blockIdx.y * 16;
    for (int i = tid; i < 16 * Kpad; i += 256) {
        int row = i / Kpad, k = i - row * Kpad;
        wlds[i] = (k < K) ? (f16)w[(ntile + row) * K + k] : (f16)0.f;
    }
    __syncthreads();

    const int wave  = tid >> 5, lane = tid & 31;
    const int half_ = lane >> 4, l15 = lane & 15;
    const long mtile = ((long)blockIdx.x * 8 + wave) * 16;

    // per-lane A row base inside the padded tensor (constant over K loop)
    const long m  = mtile + l15;
    const int bb  = (int)(m / spatial);
    int rem       = (int)(m - (long)bb * spatial);
    const int od  = rem / (H * W);  rem -= od * (H * W);
    const int oh  = rem / W;
    const int ow  = rem - oh * W;
    const long base = (long)bb * IC * Vol + od * HpWp + oh * Wp + ow;

    v8f acc = {};
#pragma unroll
    for (int kb = 0; kb < Kpad; kb += 32) {
        v16h a, bf;
#pragma unroll
        for (int e = 0; e < 16; ++e) {
            // 16-bit A 16x32 layout: lanes0-15 -> K {0..7,16..23}, lanes16-31 -> +8
            const int kc = kb + (e < 8 ? e : e + 8);
            f16 v = (f16)0.f;
            if (kc + 8 < K) {               // both halves valid (compile-time)
                int off = half_ ? koff_of(kc + 8, HpWp, Wp, Vol)
                                : koff_of(kc,     HpWp, Wp, Vol);
                v = xpad[base + off];
            } else if (kc < K) {            // hi half falls into K padding
                f16 tv = xpad[base + koff_of(kc, HpWp, Wp, Vol)];
                v = half_ ? (f16)0.f : tv;
            }
            a[e] = v;
        }
        // B 32x16 layout: col = l15, element e -> K = kb + half*8 + (e < 8 ? e : e + 8)
        const f16* wr = &wlds[l15 * Kpad + kb + half_ * 8];
#pragma unroll
        for (int e = 0; e < 16; ++e) bf[e] = wr[e < 8 ? e : e + 8];
        acc = __builtin_amdgcn_wmma_f32_16x16x32_f16(false, a, false, bf,
                                                     (short)0, acc, false, false);
    }

    const int n    = ntile + l15;
    const float bv = bias[n];
#pragma unroll
    for (int r = 0; r < 8; ++r) {
        long mo = mtile + r + half_ * 8;      // C/D layout
        int b2  = (int)(mo / spatial);
        int r2  = (int)(mo - (long)b2 * spatial);
        float v = acc[r] + bv;
        v = v > 0.f ? v : 0.f;                // fused ReLU
        y[((long)b2 * OC + n) * spatial + r2] = v;   // NCDHW f32
    }
}

// zero-fill an f16 buffer (halo init for padded tensors)
__global__ __launch_bounds__(256) void zero_h_k(f16* __restrict__ p, long n)
{
    long i = (long)blockIdx.x * blockDim.x + threadIdx.x;
    if (i < n) p[i] = (f16)0.f;
}

// pack f32 NCDHW tensor into padded f16 (interior only; halo pre-zeroed)
__global__ __launch_bounds__(256) void pad_x_k(
    const float* __restrict__ x, f16* __restrict__ xpad,
    long total, int D, int H, int W)
{
    long n = (long)blockIdx.x * blockDim.x + threadIdx.x;
    if (n >= total) return;
    int w = (int)(n % W); long t = n / W;
    int h = (int)(t % H); t /= H;
    int d = (int)(t % D); t /= D;             // t = b*IC + c
    long dst = ((t * (D + 2) + d + 1) * (long)((H + 2) * (W + 2)))
             + (h + 1) * (W + 2) + (w + 1);
    xpad[dst] = (f16)x[n];
}

// fused MaxPool3d (2,2,1) + write into next stage's padded f16 input
__global__ __launch_bounds__(256) void pool_pad_k(
    const float* __restrict__ x, f16* __restrict__ ypad,
    long total, int D, int H, int W)
{
    const int OD = D >> 1, OH = H >> 1, OW = W;
    long n = (long)blockIdx.x * blockDim.x + threadIdx.x;
    if (n >= total) return;
    int ow = (int)(n % OW); long t = n / OW;
    int oh = (int)(t % OH); t /= OH;
    int od = (int)(t % OD); t /= OD;          // t = b*C + c
    const float* xp = x + (t * D + od * 2) * (long)(H * W) + (oh * 2) * W + ow;
    float v = fmaxf(fmaxf(xp[0], xp[W]), fmaxf(xp[(long)H * W], xp[(long)H * W + W]));
    long dst = ((t * (OD + 2) + od + 1) * (long)((OH + 2) * (OW + 2)))
             + (oh + 1) * (OW + 2) + (ow + 1);
    ypad[dst] = (f16)v;
}

// MaxPool3d (2,2,1), f32 -> f32 channel-last (B,D,H,W,C) for the flatten path
__global__ __launch_bounds__(256) void pool_cl_k(
    const float* __restrict__ x, float* __restrict__ y,
    int B, int C, int D, int H, int W)
{
    const int OD = D >> 1, OH = H >> 1, OW = W;
    long n = (long)blockIdx.x * blockDim.x + threadIdx.x;
    long total = (long)B * C * OD * OH * OW;
    if (n >= total) return;
    int ow = (int)(n % OW); long t = n / OW;
    int oh = (int)(t % OH); t /= OH;
    int od = (int)(t % OD); t /= OD;
    int c  = (int)(t % C);
    int b  = (int)(t / C);
    const float* xp = x + (((long)b * C + c) * D + od * 2) * (long)(H * W) + (oh * 2) * W + ow;
    float v = fmaxf(fmaxf(xp[0], xp[W]), fmaxf(xp[(long)H * W], xp[(long)H * W + W]));
    y[((((long)b * OD + od) * OH + oh) * OW + ow) * C + c] = v;
}

// mean over P positions of channel-last (B,P,C)
__global__ __launch_bounds__(256) void mean_k(
    const float* __restrict__ flat, float* __restrict__ mean, int B, int P, int C)
{
    int i = blockIdx.x * blockDim.x + threadIdx.x;
    if (i >= B * C) return;
    int c = i % C, b = i / C;
    const float* p = flat + (long)b * P * C + c;
    float s = 0.f;
    for (int j = 0; j < P; ++j) s += p[(long)j * C];
    mean[i] = s / (float)P;
}

// ---------------------------------------------------------------------------
// f32 GEMM: Y[m][n] = act( sum_k X[m][k]*W[n][k] + b1[n] + b2[n] )
// Block = 8 M-tiles x 1 N-tile. The 16-row weight slab for each 256-wide
// K-chunk is staged into LDS with global_load_async_to_lds_b128 (ASYNCcnt),
// shared by all 8 waves; compute on v_wmma_f32_16x16x4_f32.
// K must be a sum of 256/64 chunks (64, 256, 5184 all qualify).
// ---------------------------------------------------------------------------
template <int ACT, bool GUARD>
__global__ __launch_bounds__(256) void gemm_k(
    const float* __restrict__ X, const float* __restrict__ Wt,
    const float* __restrict__ b1, const float* __restrict__ b2,
    float* __restrict__ Y, int M, int N, int K)
{
    __shared__ float wtile[16 * 256];            // 16 KB weight slab
    const int tid = threadIdx.x;
    const int wave = tid >> 5, lane = tid & 31;
    const int half_ = lane >> 4, l15 = lane & 15;
    const int nbase = blockIdx.y * 16;
    const long mtile = ((long)blockIdx.x * 8 + wave) * 16;
    const int am = (int)mtile + l15;             // M always multiple of 16 here
    const float* xrow = X + (long)am * K;
    __builtin_prefetch(xrow, 0, 3);              // global_prefetch_b8

    v8f acc = {};
    for (int kb0 = 0; kb0 < K; kb0 += 256) {
        const int chunk  = (K - kb0 < 256) ? (K - kb0) : 256;   // 64 or 256
        const int lshift = (chunk == 256) ? 8 : 6;
        // cooperative async stage: 256 threads x 16B; iters = 16*chunk/1024
        const int iters = (16 * chunk) >> 10;
        for (int it = 0; it < iters; ++it) {
            int idx = (it * 256 + tid) * 4;      // float index within slab
            int row = idx >> lshift;
            int col = idx & (chunk - 1);
            int wr  = GUARD ? ((nbase + row < N) ? nbase + row : 0) : nbase + row;
            const float* g = Wt + (long)wr * K + kb0 + col;
            unsigned ldsb  = (unsigned)(size_t)&wtile[row * 256 + col];
            asm volatile("global_load_async_to_lds_b128 %0, %1, off"
                         :: "v"(ldsb), "v"(g) : "memory");
        }
        asm volatile("s_wait_asynccnt 0x0" ::: "memory");
        __syncthreads();                          // slab visible to all waves

#pragma unroll 8
        for (int kb = 0; kb < chunk; kb += 4) {
            v2f a, bf;
#pragma unroll
            for (int e = 0; e < 2; ++e) {
                int kk = kb + half_ * 2 + e;      // f32 A 16x4 layout
                a[e]  = xrow[kb0 + kk];
                bf[e] = wtile[l15 * 256 + kk];
            }
            acc = __builtin_amdgcn_wmma_f32_16x16x4_f32(false, a, false, bf,
                                                        (short)0, acc, false, false);
        }
        __syncthreads();                          // all reads done before re-stage
    }

    const int bn = nbase + l15;
    float badd = 0.f;
    if (!GUARD || bn < N) { if (b1) badd += b1[bn]; if (b2) badd += b2[bn]; }
#pragma unroll
    for (int r = 0; r < 8; ++r) {
        long mo = mtile + r + half_ * 8;
        int  no = nbase + l15;
        if (!GUARD || no < N) {
            float v = acc[r] + badd;
            if (ACT == 1) v = tanhf(v);
            Y[mo * N + no] = v;
        }
    }
}

// ---------------------------------------------------------------------------
// Fused RNN step: block owns 64 rows; both layers + prediction per launch.
// ---------------------------------------------------------------------------
__global__ __launch_bounds__(256) void rnn_step_k(
    const float* __restrict__ ihc,
    const float* __restrict__ whh1, const float* __restrict__ wih2,
    const float* __restrict__ whh2,
    const float* __restrict__ bih2, const float* __restrict__ bhh2,
    const float* __restrict__ wfc,  const float* __restrict__ bfc,
    float* __restrict__ h, float* __restrict__ h2,
    float* __restrict__ h2n, float* __restrict__ preds, int t)
{
    __shared__ float hn[64 * 256];               // 64 KB
    const int tid = threadIdx.x;
    const int wave = tid >> 5, lane = tid & 31;
    const int half_ = lane >> 4, l15 = lane & 15;
    const int rowbase = blockIdx.x * 64;

    // phase 1: hn = tanh(ihc + h @ whh1^T)
    for (int tile = wave; tile < 64; tile += 8) {
        int mt = tile >> 4, nt = tile & 15;
        int mb = mt * 16, nb = nt * 16;
        const float* hrow = h    + (long)(rowbase + mb + l15) * 256;
        const float* wrow = whh1 + (long)(nb + l15) * 256;
        __builtin_prefetch(wrow, 0, 3);
        v8f acc = {};
#pragma unroll 8
        for (int kb = 0; kb < 256; kb += 4) {
            v2f a, bf;
#pragma unroll
            for (int e = 0; e < 2; ++e) {
                int k = kb + half_ * 2 + e;
                a[e]  = hrow[k];
                bf[e] = wrow[k];
            }
            acc = __builtin_amdgcn_wmma_f32_16x16x4_f32(false, a, false, bf,
                                                        (short)0, acc, false, false);
        }
#pragma unroll
        for (int r = 0; r < 8; ++r) {
            int lr  = mb + r + half_ * 8;
            int col = nb + l15;
            hn[lr * 256 + col] = tanhf(acc[r] + ihc[(long)(rowbase + lr) * 256 + col]);
        }
    }
    __syncthreads();

    // phase 2: h2n = tanh(hn @ wih2^T + h2 @ whh2^T + bih2 + bhh2)
    for (int tile = wave; tile < 64; tile += 8) {
        int mt = tile >> 4, nt = tile & 15;
        int mb = mt * 16, nb = nt * 16;
        const float* arow  = &hn[(mb + l15) * 256];
        const float* w1row = wih2 + (long)(nb + l15) * 256;
        const float* h2row = h2   + (long)(rowbase + mb + l15) * 256;
        const float* w2row = whh2 + (long)(nb + l15) * 256;
        __builtin_prefetch(w1row, 0, 3);
        __builtin_prefetch(w2row, 0, 3);
        v8f acc = {};
#pragma unroll 8
        for (int kb = 0; kb < 256; kb += 4) {
            v2f a, bf;
#pragma unroll
            for (int e = 0; e < 2; ++e) {
                int k = kb + half_ * 2 + e;
                a[e]  = arow[k];
                bf[e] = w1row[k];
            }
            acc = __builtin_amdgcn_wmma_f32_16x16x4_f32(false, a, false, bf,
                                                        (short)0, acc, false, false);
        }
#pragma unroll 8
        for (int kb = 0; kb < 256; kb += 4) {
            v2f a, bf;
#pragma unroll
            for (int e = 0; e < 2; ++e) {
                int k = kb + half_ * 2 + e;
                a[e]  = h2row[k];
                bf[e] = w2row[k];
            }
            acc = __builtin_amdgcn_wmma_f32_16x16x4_f32(false, a, false, bf,
                                                        (short)0, acc, false, false);
        }
#pragma unroll
        for (int r = 0; r < 8; ++r) {
            int lr  = mb + r + half_ * 8;
            int col = nb + l15;
            h2n[(long)(rowbase + lr) * 256 + col] = tanhf(acc[r] + bih2[col] + bhh2[col]);
        }
    }
    __threadfence_block();
    __syncthreads();

    // phase 3: commit state + prediction
    for (int i = tid; i < 64 * 256; i += 256) {
        int lr = i >> 8, col = i & 255;
        long g = (long)(rowbase + lr) * 256 + col;
        h[g]  = hn[i];
        h2[g] = h2n[g];
    }
    if (tid < 64) {
        int row = rowbase + tid;
        const float* hp = h2n + (long)row * 256;
        float s = bfc[0];
        for (int c = 0; c < 256; ++c) s += hp[c] * wfc[c];
        preds[row * 23 + t] = s;
    }
}

// ---------------------------------------------------------------------------
extern "C" void kernel_launch(void* const* d_in, const int* in_sizes, int n_in,
                              void* d_out, int out_size, void* d_ws, size_t ws_size,
                              hipStream_t stream)
{
    (void)in_sizes; (void)n_in; (void)out_size; (void)ws_size;
    const float* x        = (const float*)d_in[0];
    const float* w1       = (const float*)d_in[1];
    const float* b1       = (const float*)d_in[2];
    const float* w2       = (const float*)d_in[3];
    const float* b2       = (const float*)d_in[4];
    const float* w3       = (const float*)d_in[5];
    const float* b3       = (const float*)d_in[6];
    const float* wih1     = (const float*)d_in[7];
    const float* whh1     = (const float*)d_in[8];
    const float* bih1     = (const float*)d_in[9];
    const float* bhh1     = (const float*)d_in[10];
    const float* wih2     = (const float*)d_in[11];
    const float* whh2     = (const float*)d_in[12];
    const float* bih2     = (const float*)d_in[13];
    const float* bhh2     = (const float*)d_in[14];
    const float* w_init_h = (const float*)d_in[15];
    const float* b_init_h = (const float*)d_in[16];
    const float* w_init_h2= (const float*)d_in[17];
    const float* b_init_h2= (const float*)d_in[18];
    const float* w_fc     = (const float*)d_in[19];
    const float* b_fc     = (const float*)d_in[20];
    const float* w_lin1   = (const float*)d_in[21];
    const float* b_lin1   = (const float*)d_in[22];
    const float* w_lin2   = (const float*)d_in[23];
    const float* b_lin2   = (const float*)d_in[24];
    float* out = (float*)d_out;

    // Workspace layout (floats)
    float* ws    = (float*)d_ws;
    float* bufA  = ws;                          // 42,467,328  conv outputs f32 (reused)
    f16*   padH  = (f16*)(bufA + 42467328L);    // 29,552,640 halfs padded f16 inputs (reused)
    float* flatC = bufA + 42467328L + 14776320L;//  2,654,208  (B,81,64) channel-last
    float* meanE = flatC + 2654208L;            //     32,768
    float* h     = meanE + 32768L;              //    131,072
    float* h2    = h     + 131072L;
    float* h2n   = h2    + 131072L;
    float* ihc   = h2n   + 131072L;
    float* hid   = ihc   + 131072L;             //    131,072

    const int B = 512;
    auto blks = [](long n) { return (unsigned)((n + 255) / 256); };

    // stage 1: pad x -> f16 (512,6,26,74,5), conv1 -> bufA (512,16,24,72,3)
    { long pz = (long)B * 6 * 26 * 74 * 5;
      zero_h_k<<<blks(pz), 256, 0, stream>>>(padH, pz);
      long ti = (long)B * 6 * 24 * 72 * 3;
      pad_x_k<<<blks(ti), 256, 0, stream>>>(x, padH, ti, 24, 72, 3);
      dim3 g((unsigned)((long)B * 24 * 72 * 3 / 16 / 8), 1);
      conv3d_k<6, 16, 24, 72, 3><<<g, 256, 0, stream>>>(padH, w1, b1, bufA, B); }
    // stage 2: pool1+pad -> f16 (512,16,14,38,5), conv2 -> bufA (512,32,12,36,3)
    { long pz = (long)B * 16 * 14 * 38 * 5;
      zero_h_k<<<blks(pz), 256, 0, stream>>>(padH, pz);
      long to = (long)B * 16 * 12 * 36 * 3;
      pool_pad_k<<<blks(to), 256, 0, stream>>>(bufA, padH, to, 24, 72, 3);
      dim3 g((unsigned)((long)B * 12 * 36 * 3 / 16 / 8), 2);
      conv3d_k<16, 32, 12, 36, 3><<<g, 256, 0, stream>>>(padH, w2, b2, bufA, B); }
    // stage 3: pool2+pad -> f16 (512,32,8,20,5), conv3 -> bufA (512,64,6,18,3)
    { long pz = (long)B * 32 * 8 * 20 * 5;
      zero_h_k<<<blks(pz), 256, 0, stream>>>(padH, pz);
      long to = (long)B * 32 * 6 * 18 * 3;
      pool_pad_k<<<blks(to), 256, 0, stream>>>(bufA, padH, to, 12, 36, 3);
      dim3 g((unsigned)((long)B * 6 * 18 * 3 / 16 / 8), 4);
      conv3d_k<32, 64, 6, 18, 3><<<g, 256, 0, stream>>>(padH, w3, b3, bufA, B); }
    // pool3, channel-last -> flat (512, 81, 64) == (512, 5184)
    { long to = (long)B * 64 * 3 * 9 * 3;
      pool_cl_k<<<blks(to), 256, 0, stream>>>(bufA, flatC, B, 64, 6, 18, 3); }
    // mean_enc (512,64)
    mean_k<<<(B * 64 + 255) / 256, 256, 0, stream>>>(flatC, meanE, B, 81, 64);

    // head GEMMs: grid = (M/128, N/16 tiles)
    // h = mean_enc @ w_init_h^T + b_init_h
    gemm_k<0, false><<<dim3(4, 16), 256, 0, stream>>>(meanE, w_init_h, b_init_h, nullptr, h, 512, 256, 64);
    // h2 = h @ w_init_h2^T + b_init_h2
    gemm_k<0, false><<<dim3(4, 16), 256, 0, stream>>>(h, w_init_h2, b_init_h2, nullptr, h2, 512, 256, 256);
    // ih_const = mean_enc @ wih1^T + bih1 + bhh1
    gemm_k<0, false><<<dim3(4, 16), 256, 0, stream>>>(meanE, wih1, bih1, bhh1, ihc, 512, 256, 64);

    // 23 fused RNN steps; preds -> out[0 .. 512*23)
    for (int t = 0; t < 23; ++t)
        rnn_step_k<<<8, 256, 0, stream>>>(ihc, whh1, wih2, whh2, bih2, bhh2, w_fc, b_fc,
                                          h, h2, h2n, out, t);

    // hid = tanh(flat @ w_lin1^T + b_lin1)
    gemm_k<1, false><<<dim3(4, 16), 256, 0, stream>>>(flatC, w_lin1, b_lin1, nullptr, hid, 512, 256, 5184);
    // out_c = hid @ w_lin2^T + b_lin2  -> out[512*23 ..)  (N=12: guarded tile)
    gemm_k<0, true><<<dim3(4, 1), 256, 0, stream>>>(hid, w_lin2, b_lin2, nullptr, out + 512 * 23, 512, 12, 256);
}
